// cluster_mil_17394617549076
// MI455X (gfx1250) — matrix operand order repaired and checked by reading, not compiled, not run on hardware
//
#include <hip/hip_runtime.h>
#include <hip/hip_bf16.h>

// Shapes (compile-time, from the reference)
#define BB   8
#define NN   2048
#define FF   1024
#define ZZ   256
#define KK   10
#define OUTD 2
#define EPSF 1e-5f

typedef __attribute__((ext_vector_type(16))) __bf16 v16bf;
typedef __attribute__((ext_vector_type(8)))  float  v8f;
typedef __attribute__((ext_vector_type(4)))  float  v4f;

// ---------------------------------------------------------------------------
// Transpose + convert f32 [rows,cols] -> bf16 [cols,rows]
// ---------------------------------------------------------------------------
__global__ __launch_bounds__(256)
void transpose_cvt_bf16(const float* __restrict__ in, __bf16* __restrict__ out,
                        int rows, int cols)
{
    int idx = blockIdx.x * blockDim.x + threadIdx.x;
    if (idx >= rows * cols) return;
    int r = idx / cols, c = idx % cols;
    out[(size_t)c * rows + r] = (__bf16)in[(size_t)r * cols + c];
}

// A-fragment loader: 16x32 A fragment in contiguous k order (as B). Lane (l16,half) loads
// K = k0+16*half+[0..15]: one contiguous f32 run,
// converted in-register to bf16 (v_cvt_pk_bf16_f32 pairs).
__device__ __forceinline__ v16bf load_a_frag(const float* __restrict__ aRow,
                                             int k0, int half)
{
    v4f a0 = *(const v4f*)(aRow + k0 + 16 * half);
    v4f a1 = *(const v4f*)(aRow + k0 + 16 * half + 4);
    v4f a2 = *(const v4f*)(aRow + k0 + 16 * half + 8);
    v4f a3 = *(const v4f*)(aRow + k0 + 16 * half + 12);
    v16bf a;
#pragma unroll
    for (int i = 0; i < 4; ++i) {
        a[i]      = (__bf16)a0[i];
        a[4 + i]  = (__bf16)a1[i];
        a[8 + i]  = (__bf16)a2[i];
        a[12 + i] = (__bf16)a3[i];
    }
    return a;
}

// ---------------------------------------------------------------------------
// WMMA GEMM: Out[M,Ncols] = A[M,Kdim] (f32, cvt->bf16) @ Bt[Ncols,Kdim]^T (bf16)
// One wave computes a 32x64 tile (2 A-frags x 4 B-frags -> 8 accumulators),
// so each B fragment is reused twice. Block = 128 thr = 4 waves = 128 rows.
// mode 1: fused ReLU(x + bias) then BatchNorm (eval) epilogue.
// ---------------------------------------------------------------------------
__global__ __launch_bounds__(128)
void wmma_gemm_kernel(const float* __restrict__ A, const __bf16* __restrict__ Bt,
                      float* __restrict__ Out, int Kdim, int Ncols,
                      const float* __restrict__ bias,
                      const float* __restrict__ gamma, const float* __restrict__ beta,
                      const float* __restrict__ mean,  const float* __restrict__ var,
                      int mode)
{
    const int lane = threadIdx.x & 31;
    const int wid  = threadIdx.x >> 5;
    const int half = lane >> 4;        // 0 or 1
    const int l16  = lane & 15;
    const int rowTile = blockIdx.x * 4 + wid;   // 32-row tile index
    const int colBase = blockIdx.y * 64;

    v8f acc[2][4] = {};

    const float*  aRow0 = A  + (size_t)(rowTile * 32 + l16) * Kdim;
    const float*  aRow1 = aRow0 + (size_t)16 * Kdim;
    const __bf16* bRow  = Bt + (size_t)(colBase + l16) * Kdim + 16 * half;

    for (int k0 = 0; k0 < Kdim; k0 += 32) {
        __builtin_prefetch(aRow0 + k0 + 64, 0, 0);   // speculative; OOB dropped
        __builtin_prefetch(aRow1 + k0 + 64, 0, 0);
        v16bf a0 = load_a_frag(aRow0, k0, half);
        v16bf a1 = load_a_frag(aRow1, k0, half);
        // B fragments: lane N = colTile+l16, K = k0+16*half+[0..15] contiguous
#pragma unroll
        for (int t = 0; t < 4; ++t) {
            v16bf b = *(const v16bf*)(bRow + (size_t)(16 * t) * Kdim + k0);
            acc[0][t] = __builtin_amdgcn_wmma_f32_16x16x32_bf16(
                false, a0, false, b, (short)0, acc[0][t], false, false);
            acc[1][t] = __builtin_amdgcn_wmma_f32_16x16x32_bf16(
                false, a1, false, b, (short)0, acc[1][t], false, false);
        }
    }

    // D layout: element v -> row = base + v + 8*half, col = colTile + l16
#pragma unroll
    for (int t = 0; t < 4; ++t) {
        const int col = colBase + 16 * t + l16;
        float bsc = 1.f, bsh = 0.f, bb = 0.f;
        if (mode == 1) {
            bb = bias[col];
            float rs = rsqrtf(var[col] + EPSF);
            bsc = gamma[col] * rs;
            bsh = beta[col] - mean[col] * bsc;
        }
#pragma unroll
        for (int mt = 0; mt < 2; ++mt) {
#pragma unroll
            for (int v = 0; v < 8; ++v) {
                float x = acc[mt][t][v];
                if (mode == 1) x = fmaxf(x + bb, 0.f) * bsc + bsh;
                Out[(size_t)(rowTile * 32 + mt * 16 + v + 8 * half) * Ncols + col] = x;
            }
        }
    }
}

// ---------------------------------------------------------------------------
// Soft assignment: per row (b,n): dist to K centroids, softmax(-dist) over K.
// One wave per row; 8 waves per block. Centroids staged in LDS.
// ---------------------------------------------------------------------------
__global__ __launch_bounds__(256)
void soft_assign_kernel(const float* __restrict__ emb, const float* __restrict__ cent,
                        float* __restrict__ soft)
{
    __shared__ float scs[KK][ZZ];
    __shared__ float cc[KK];
    for (int i = threadIdx.x; i < KK * ZZ; i += 256)
        scs[i / ZZ][i % ZZ] = cent[i];
    __syncthreads();
    if (threadIdx.x < KK) {
        float s = 0.f;
        for (int z = 0; z < ZZ; ++z) { float v = scs[threadIdx.x][z]; s += v * v; }
        cc[threadIdx.x] = s;
    }
    __syncthreads();

    const int lane = threadIdx.x & 31;
    const int wid  = threadIdx.x >> 5;
    const int row  = blockIdx.x * 8 + wid;          // over B*N
    const float* e = emb + (size_t)row * ZZ;

    float ee = 0.f;
    float ck[KK];
#pragma unroll
    for (int k = 0; k < KK; ++k) ck[k] = 0.f;
    for (int z = lane; z < ZZ; z += 32) {
        float v = e[z];
        ee += v * v;
#pragma unroll
        for (int k = 0; k < KK; ++k) ck[k] = fmaf(v, scs[k][z], ck[k]);
    }
    for (int o = 16; o > 0; o >>= 1) {
        ee += __shfl_xor(ee, o, 32);
#pragma unroll
        for (int k = 0; k < KK; ++k) ck[k] += __shfl_xor(ck[k], o, 32);
    }
    if (lane == 0) {
        float nd[KK]; float mx = -1e30f;
#pragma unroll
        for (int k = 0; k < KK; ++k) {
            float d2 = ee + cc[k] - 2.f * ck[k];
            nd[k] = -sqrtf(fmaxf(d2, 0.f));
            mx = fmaxf(mx, nd[k]);
        }
        float sum = 0.f;
#pragma unroll
        for (int k = 0; k < KK; ++k) { nd[k] = expf(nd[k] - mx); sum += nd[k]; }
        float inv = 1.f / sum;
#pragma unroll
        for (int k = 0; k < KK; ++k) soft[(size_t)row * KK + k] = nd[k] * inv;
    }
}

// ---------------------------------------------------------------------------
// Instance scores: s[b,n,k] = ia_w2 . tanh(soft[b,n,k]*e1[b,n,:] + ia_b1) + ia_b2
// One wave per (b,n,k) task; 8 waves per block.
// ---------------------------------------------------------------------------
__global__ __launch_bounds__(256)
void inst_score_kernel(const float* __restrict__ e1, const float* __restrict__ soft,
                       const float* __restrict__ b1, const float* __restrict__ w2,
                       const float* __restrict__ b2, float* __restrict__ s)
{
    const int lane = threadIdx.x & 31;
    const int wid  = threadIdx.x >> 5;
    const int task = blockIdx.x * 8 + wid;          // over B*N*K
    const int row  = task / KK;
    const float scv = soft[task];
    const float* er = e1 + (size_t)row * ZZ;
    float acc = 0.f;
    for (int z = lane; z < ZZ; z += 32)
        acc = fmaf(tanhf(fmaf(scv, er[z], b1[z])), w2[z], acc);
    for (int o = 16; o > 0; o >>= 1) acc += __shfl_xor(acc, o, 32);
    if (lane == 0) s[task] = acc + b2[0];
}

// ---------------------------------------------------------------------------
// Instance softmax over N per (b,k); writes g = attn * soft.
// ---------------------------------------------------------------------------
__global__ __launch_bounds__(256)
void inst_softmax_kernel(const float* __restrict__ s, const float* __restrict__ soft,
                         float* __restrict__ g)
{
    const int b = blockIdx.x / KK, k = blockIdx.x % KK;
    __shared__ float red[256];
    const size_t base = ((size_t)b * NN) * KK + k;

    float mx = -1e30f;
    for (int n = threadIdx.x; n < NN; n += 256)
        mx = fmaxf(mx, s[base + (size_t)n * KK]);
    red[threadIdx.x] = mx; __syncthreads();
    for (int o = 128; o > 0; o >>= 1) {
        if (threadIdx.x < o) red[threadIdx.x] = fmaxf(red[threadIdx.x], red[threadIdx.x + o]);
        __syncthreads();
    }
    mx = red[0]; __syncthreads();

    float sum = 0.f;
    for (int n = threadIdx.x; n < NN; n += 256)
        sum += expf(s[base + (size_t)n * KK] - mx);
    red[threadIdx.x] = sum; __syncthreads();
    for (int o = 128; o > 0; o >>= 1) {
        if (threadIdx.x < o) red[threadIdx.x] += red[threadIdx.x + o];
        __syncthreads();
    }
    const float inv = 1.f / red[0];

    for (int n = threadIdx.x; n < NN; n += 256) {
        size_t idx = base + (size_t)n * KK;
        g[idx] = expf(s[idx] - mx) * inv * soft[idx];
    }
}

// ---------------------------------------------------------------------------
// Cluster embeddings: cl[b,k,z] = sum_n g[b,n,k] * emb[b,n,z]
// One block per (b,k); thread = z.
// ---------------------------------------------------------------------------
__global__ __launch_bounds__(256)
void cluster_embed_kernel(const float* __restrict__ g, const float* __restrict__ emb,
                          float* __restrict__ cl)
{
    const int b = blockIdx.x / KK, k = blockIdx.x % KK;
    const int z = threadIdx.x;
    const float* eb = emb + (size_t)b * NN * ZZ;
    const float* gb = g + (size_t)b * NN * KK + k;
    float acc = 0.f;
    for (int n = 0; n < NN; ++n)
        acc = fmaf(gb[(size_t)n * KK], eb[(size_t)n * ZZ + z], acc);
    cl[((size_t)b * KK + k) * ZZ + z] = acc;
}

// ---------------------------------------------------------------------------
// Cluster attention scores: sc[b,k] = ca_w2 . tanh(cl[b,k,:]@ca_W1 + ca_b1) + ca_b2
// ---------------------------------------------------------------------------
__global__ __launch_bounds__(256)
void cluster_score_kernel(const float* __restrict__ cl, const float* __restrict__ W1,
                          const float* __restrict__ b1, const float* __restrict__ w2,
                          const float* __restrict__ b2, float* __restrict__ sc)
{
    __shared__ float scl[ZZ];
    __shared__ float red[256];
    const int bk = blockIdx.x;
    const int d = threadIdx.x;
    scl[d] = cl[(size_t)bk * ZZ + d];
    __syncthreads();
    float acc = b1[d];
    for (int z = 0; z < ZZ; ++z)
        acc = fmaf(scl[z], W1[(size_t)z * ZZ + d], acc);
    red[d] = tanhf(acc) * w2[d];
    __syncthreads();
    for (int o = 128; o > 0; o >>= 1) {
        if (d < o) red[d] += red[d + o];
        __syncthreads();
    }
    if (d == 0) sc[bk] = red[0] + b2[0];
}

// ---------------------------------------------------------------------------
// Head: ca = softmax_K(sc); pooled = sum_k ca*cl; BN; out = pooled@head_W + b
// One block per bag; thread = z.
// ---------------------------------------------------------------------------
__global__ __launch_bounds__(256)
void head_kernel(const float* __restrict__ sc, const float* __restrict__ cl,
                 const float* __restrict__ hg, const float* __restrict__ hb,
                 const float* __restrict__ hm, const float* __restrict__ hv,
                 const float* __restrict__ hW, const float* __restrict__ hbias,
                 float* __restrict__ out)
{
    const int b = blockIdx.x;
    __shared__ float ca[KK];
    __shared__ float red[256];
    if (threadIdx.x == 0) {
        float mx = -1e30f;
        for (int k = 0; k < KK; ++k) mx = fmaxf(mx, sc[b * KK + k]);
        float sum = 0.f;
        for (int k = 0; k < KK; ++k) { ca[k] = expf(sc[b * KK + k] - mx); sum += ca[k]; }
        for (int k = 0; k < KK; ++k) ca[k] /= sum;
    }
    __syncthreads();
    const int z = threadIdx.x;
    float p = 0.f;
#pragma unroll
    for (int k = 0; k < KK; ++k)
        p = fmaf(ca[k], cl[((size_t)b * KK + k) * ZZ + z], p);
    float pb = (p - hm[z]) * rsqrtf(hv[z] + EPSF) * hg[z] + hb[z];
    for (int o = 0; o < OUTD; ++o) {
        red[z] = pb * hW[(size_t)z * OUTD + o];
        __syncthreads();
        for (int s2 = 128; s2 > 0; s2 >>= 1) {
            if (z < s2) red[z] += red[z + s2];
            __syncthreads();
        }
        if (z == 0) out[b * OUTD + o] = red[0] + hbias[o];
        __syncthreads();
    }
}

// ---------------------------------------------------------------------------
extern "C" void kernel_launch(void* const* d_in, const int* in_sizes, int n_in,
                              void* d_out, int out_size, void* d_ws, size_t ws_size,
                              hipStream_t stream)
{
    const float* bags   = (const float*)d_in[0];
    const float* enc_W  = (const float*)d_in[1];
    const float* enc_b  = (const float*)d_in[2];
    const float* bn1_g  = (const float*)d_in[3];
    const float* bn1_b  = (const float*)d_in[4];
    const float* bn1_m  = (const float*)d_in[5];
    const float* bn1_v  = (const float*)d_in[6];
    const float* cent   = (const float*)d_in[7];
    const float* ia_W1  = (const float*)d_in[8];
    const float* ia_b1  = (const float*)d_in[9];
    const float* ia_w2  = (const float*)d_in[10];
    const float* ia_b2  = (const float*)d_in[11];
    const float* ca_W1  = (const float*)d_in[12];
    const float* ca_b1  = (const float*)d_in[13];
    const float* ca_w2  = (const float*)d_in[14];
    const float* ca_b2  = (const float*)d_in[15];
    const float* h_g    = (const float*)d_in[16];
    const float* h_b    = (const float*)d_in[17];
    const float* h_m    = (const float*)d_in[18];
    const float* h_v    = (const float*)d_in[19];
    const float* h_W    = (const float*)d_in[20];
    const float* h_bias = (const float*)d_in[21];
    float* out = (float*)d_out;
    (void)in_sizes; (void)n_in; (void)out_size; (void)ws_size;

    char* wsb = (char*)d_ws;
    size_t off = 0;
    auto carve = [&](size_t bytes) -> void* {
        size_t cur = (off + 255) & ~(size_t)255;
        off = cur + bytes;
        return (void*)(wsb + cur);
    };
    __bf16* encWt = (__bf16*)carve((size_t)ZZ * FF * sizeof(__bf16));
    __bf16* iaW1t = (__bf16*)carve((size_t)ZZ * ZZ * sizeof(__bf16));
    float* emb  = (float*)carve((size_t)BB * NN * ZZ * sizeof(float));
    float* e1   = (float*)carve((size_t)BB * NN * ZZ * sizeof(float));
    float* soft = (float*)carve((size_t)BB * NN * KK * sizeof(float));
    float* sbuf = (float*)carve((size_t)BB * NN * KK * sizeof(float));
    float* gbuf = (float*)carve((size_t)BB * NN * KK * sizeof(float));
    float* cl   = (float*)carve((size_t)BB * KK * ZZ * sizeof(float));
    float* scb  = (float*)carve((size_t)BB * KK * sizeof(float));

    // Weight transposes -> bf16 (so B-fragments are contiguous 32B loads)
    transpose_cvt_bf16<<<(FF * ZZ) / 256, 256, 0, stream>>>(enc_W, encWt, FF, ZZ);
    transpose_cvt_bf16<<<(ZZ * ZZ) / 256, 256, 0, stream>>>(ia_W1, iaW1t, ZZ, ZZ);

    // GEMM 1: emb = BN(ReLU(bags @ enc_W + enc_b))   [16384 x 1024 x 256]
    dim3 gg((BB * NN) / 128, ZZ / 64);
    wmma_gemm_kernel<<<gg, 128, 0, stream>>>(bags, encWt, emb, FF, ZZ,
                                             enc_b, bn1_g, bn1_b, bn1_m, bn1_v, 1);

    // Soft cluster assignment (softmax over -cdist)
    soft_assign_kernel<<<(BB * NN) / 8, 256, 0, stream>>>(emb, cent, soft);

    // GEMM 2: e1 = emb @ ia_W1                        [16384 x 256 x 256]
    wmma_gemm_kernel<<<gg, 128, 0, stream>>>(emb, iaW1t, e1, ZZ, ZZ,
                                             nullptr, nullptr, nullptr, nullptr, nullptr, 0);

    // s[b,n,k], attn softmax over N, g = attn*soft
    inst_score_kernel<<<(BB * NN * KK) / 8, 256, 0, stream>>>(e1, soft, ia_b1, ia_w2, ia_b2, sbuf);
    inst_softmax_kernel<<<BB * KK, 256, 0, stream>>>(sbuf, soft, gbuf);

    // cl[b,k,z] = sum_n g * emb
    cluster_embed_kernel<<<BB * KK, 256, 0, stream>>>(gbuf, emb, cl);

    // cluster attention + head
    cluster_score_kernel<<<BB * KK, 256, 0, stream>>>(cl, ca_W1, ca_b1, ca_w2, ca_b2, scb);
    head_kernel<<<BB, 256, 0, stream>>>(scb, cl, h_g, h_b, h_m, h_v, h_W, h_bias, out);
}
